// GraphReduction_50130858279608
// MI455X (gfx1250) — matrix operand order repaired and checked by reading, not compiled, not run on hardware
//
#include <hip/hip_runtime.h>
#include <hip/hip_bf16.h>
#include <math.h>

// ---------------- problem constants ----------------
#define G_   512
#define N0_  64
#define INC_ 128
#define HID_ 64
#define OUTC_ 64
#define K1_  52
#define K2_  42
#define K3_  34
#define EPG_ 1024

// padded leading dims for bf16 LDS tiles (rows stay 16B-aligned; 144B/272B row
// stride vs 256B bank period spreads lanes across the 64 LDS banks)
#define LDP64  72
#define LDP128 136

typedef __attribute__((ext_vector_type(16))) __bf16 v16bf;
typedef __attribute__((ext_vector_type(8)))  __bf16 v8bf;
typedef __attribute__((ext_vector_type(8)))  float  v8f;

// ---------------- WMMA fragment helpers (wave32 layouts, CDNA5 ISA 7.12.2) ----
// A-matrix 16x32 bf16, row-major MxK in LDS. Per lane: row = m0 + L%16,
// elements 0..7 -> K = k0 + (L/16)*8 + e ; elements 8..15 -> +16.
// Two contiguous 16B runs -> 2x ds_load_b128.
__device__ __forceinline__ v16bf load_a_frag(const __bf16* t, int ld, int m0, int k0, int lane) {
    int base = (m0 + (lane & 15)) * ld + k0 + ((lane >> 4) << 3);
    v8bf lo = *(const v8bf*)(t + base);
    v8bf hi = *(const v8bf*)(t + base + 16);
    return __builtin_shufflevector(lo, hi, 0,1,2,3,4,5,6,7,8,9,10,11,12,13,14,15);
}
// B-matrix 32x16 bf16 stored TRANSPOSED (NxK) in LDS. Per lane: col = n0+L%16,
// elements 0..7 -> K = k0 + (L/16)*8 + e ; elements 8..15 -> +16 -> 2x ds_load_b128.
__device__ __forceinline__ v16bf load_bT_frag(const __bf16* t, int ld, int k0, int n0, int lane) {
    int base = (n0 + (lane & 15)) * ld + k0 + ((lane >> 4) << 3);
    v8bf lo = *(const v8bf*)(t + base);
    v8bf hi = *(const v8bf*)(t + base + 16);
    return __builtin_shufflevector(lo, hi, 0,1,2,3,4,5,6,7,8,9,10,11,12,13,14,15);
}
// C/D 16x16 f32: VGPR r -> M = r + (L/16)*8, N = L%16
__device__ __forceinline__ void store_c_frag(float* t, int ld, int m0, int n0, int lane, v8f c) {
    int col  = n0 + (lane & 15);
    int half = lane >> 4;
#pragma unroll
    for (int r = 0; r < 8; ++r)
        t[(m0 + half * 8 + r) * ld + col] = c[r];
}

// C(MxN,f32,LDS) = A(MxK,bf16,LDS,row-major) @ B, with B given TRANSPOSED
// (BT is NxK bf16 in LDS). Tile loop is wave-uniform -> EXEC all ones at WMMA.
__device__ __forceinline__ void mm_bf16(const __bf16* A, int lda, const __bf16* BT, int ldbt,
                                        float* C, int ldc, int M, int N, int K,
                                        int wave, int nwaves, int lane) {
    int tm = M >> 4, tn = N >> 4;
    for (int t = wave; t < tm * tn; t += nwaves) {
        int ti = t / tn, tj = t % tn;
        v8f acc = {};
        for (int k0 = 0; k0 < K; k0 += 32) {
            v16bf a = load_a_frag(A, lda, ti * 16, k0, lane);
            v16bf b = load_bT_frag(BT, ldbt, k0, tj * 16, lane);
            acc = __builtin_amdgcn_wmma_f32_16x16x32_bf16(
                false, a, false, b, (short)0, acc, false, false);
        }
        store_c_frag(C, ldc, ti * 16, tj * 16, lane, acc);
    }
}

// ---------------- kernels ----------------
__global__ void zero_f32(float* __restrict__ p, int n) {
    int i = blockIdx.x * blockDim.x + threadIdx.x;
    if (i < n) p[i] = 0.f;
}

__global__ void scatter_adj(const int* __restrict__ ei, float* __restrict__ A, int E) {
    int e = blockIdx.x * blockDim.x + threadIdx.x;
    if (e >= E) return;
    int src = ei[e], dst = ei[E + e];
    int g = dst >> 6;
    A[(size_t)g * 4096 + (size_t)(dst & 63) * 64 + (src & 63)] = 1.0f;
}

// h = relu( Dhat^-1/2 * A1 * Dhat^-1/2 @ (x @ W) + b ), padded to 64 rows
__global__ __launch_bounds__(256) void gcn_kernel(
    const float* __restrict__ x, int in_c,
    const float* __restrict__ A, int binarize,
    const float* __restrict__ W, const float* __restrict__ b,
    float* __restrict__ hout, int n)
{
    __shared__ float  Af[64 * 64];
    __shared__ __align__(16) __bf16 Ahbf[64 * LDP64];     // A operand (row-major)
    __shared__ __align__(16) __bf16 Xbf[64 * LDP128];     // A operand (row-major)
    __shared__ __align__(16) __bf16 Wt[64 * LDP128];      // B transposed: Wt[n][k]
    __shared__ float  XWf[64 * 64];
    __shared__ __align__(16) __bf16 XWt[64 * LDP64];      // B transposed: XWt[c][j]
    __shared__ float  dis[64];

    int g = blockIdx.x, tid = threadIdx.x;
    int lane = tid & 31, wave = tid >> 5;
    int ldx = in_c + 8;

    for (int idx = tid; idx < 4096; idx += 256) {
        int i = idx >> 6, j = idx & 63;
        float v = (i < n && j < n) ? A[(size_t)g * 4096 + idx] : 0.f;
        if (binarize) v = (v != 0.f) ? 1.f : 0.f;
        Af[idx] = v;
    }
    __syncthreads();
    if (tid < n) { float d = Af[tid * 65]; Af[tid * 65] = (d == 0.f) ? 1.f : d; }
    __syncthreads();
    if (tid < 64) {
        float s = 0.f;
        for (int j = 0; j < 64; ++j) s += Af[tid * 64 + j];
        dis[tid] = (s > 0.f) ? rsqrtf(s) : 0.f;
    }
    __syncthreads();
    for (int idx = tid; idx < 4096; idx += 256) {
        int i = idx >> 6, j = idx & 63;
        Ahbf[i * LDP64 + j] = (__bf16)(dis[i] * Af[idx] * dis[j]);
    }
    for (int idx = tid; idx < 64 * in_c; idx += 256) {
        int i = idx / in_c, c = idx - i * in_c;
        Xbf[i * ldx + c] = (__bf16)x[(size_t)g * 64 * in_c + idx];
    }
    for (int idx = tid; idx < in_c * 64; idx += 256) {
        int k = idx >> 6, nn = idx & 63;
        Wt[nn * ldx + k] = (__bf16)W[idx];
    }
    __syncthreads();
    mm_bf16(Xbf, ldx, Wt, ldx, XWf, 64, 64, 64, in_c, wave, 8, lane);
    __syncthreads();
    for (int idx = tid; idx < 4096; idx += 256) {
        int c = idx >> 6, j = idx & 63;
        XWt[c * LDP64 + j] = (__bf16)XWf[j * 64 + c];
    }
    __syncthreads();
    mm_bf16(Ahbf, LDP64, XWt, LDP64, Af, 64, 64, 64, 64, wave, 8, lane);  // reuse Af
    __syncthreads();
    for (int idx = tid; idx < 4096; idx += 256) {
        int i = idx >> 6, j = idx & 63;
        hout[(size_t)g * 4096 + idx] = (i < n) ? fmaxf(Af[idx] + b[j], 0.f) : 0.f;
    }
}

// ASAP pooling: masked max -> xq@Wlin -> attention softmax -> xn = S@x ->
// fitness -> stable top-k -> xo, An = P A1 P^T (diag zeroed), + readout accum.
__global__ __launch_bounds__(256) void pool_kernel(
    const float* __restrict__ x, const float* __restrict__ A,
    const float* __restrict__ Wlin, const float* __restrict__ blin,
    const float* __restrict__ attw, const float* __restrict__ attb_p,
    const float* __restrict__ lew1, const float* __restrict__ leb1,
    const float* __restrict__ lew2, const float* __restrict__ lew3,
    const float* __restrict__ leb3,
    float* __restrict__ hout, float* __restrict__ Aout, float* __restrict__ xs,
    int n, int k)
{
    __shared__ float  Xf[4096], A1f[4096], Qf[4096], XQf[4096], Sf[4096], XNf[4096];
    __shared__ __align__(16) __bf16 XbfT[64 * LDP64];  // B^T of x: XbfT[c][j]
    __shared__ __align__(16) __bf16 A1bf[64 * LDP64];  // A operand
    __shared__ __align__(16) __bf16 Qbf[64 * LDP64];   // A operand (xqpre)
    __shared__ __align__(16) __bf16 Wlt[64 * LDP64];   // B^T of Wlin
    __shared__ __align__(16) __bf16 Sbf[64 * LDP64];   // A operand (S)
    __shared__ float  sq[64], sx[64], v1[64], v2[64], v3[64], fitv[64], topv[64];
    __shared__ int    perm[64];
    float*  Tf   = Qf;     // xqpre(f32 scratch) dead
    float*  ANf  = XQf;    // xq dead after sq; xo dead after writeback
    __bf16* Pbf  = XbfT;   // P row-major kxn: acts as B^T for T-GEMM AND A for An-GEMM
    __bf16* TbfT = A1bf;   // T^T: TbfT[s][j]; A1bf dead after T-GEMM

    int g = blockIdx.x, tid = threadIdx.x;
    int lane = tid & 31, wave = tid >> 5;
    float ab = attb_p[0];

    // (a) load x and A1 = diag_fill(A), both padded with zeros to 64
    for (int idx = tid; idx < 4096; idx += 256) {
        int i = idx >> 6, j = idx & 63;
        float xv = x[(size_t)g * 4096 + idx];
        Xf[idx] = xv;
        XbfT[j * LDP64 + i] = (__bf16)xv;           // transposed store
        A1f[idx] = (i < n && j < n) ? A[(size_t)g * 4096 + idx] : 0.f;
    }
    __syncthreads();
    if (tid < n) { float d = A1f[tid * 65]; A1f[tid * 65] = (d == 0.f) ? 1.f : d; }
    __syncthreads();
    for (int idx = tid; idx < 4096; idx += 256) {
        int i = idx >> 6, j = idx & 63;
        A1bf[i * LDP64 + j] = (__bf16)A1f[idx];
    }

    // (b) xqpre[i,c] = max_{j: mask[i,j]} x[j,c]
    for (int idx = tid; idx < 4096; idx += 256) {
        int i = idx >> 6, c = idx & 63;
        float m = 0.f;
        if (i < n) {
            m = -3.4e38f;
            for (int j = 0; j < n; ++j)
                if (A1f[i * 64 + j] != 0.f) m = fmaxf(m, Xf[j * 64 + c]);
        }
        Qbf[i * LDP64 + c] = (__bf16)m;
    }
    for (int idx = tid; idx < 4096; idx += 256) {
        int kk = idx >> 6, nn = idx & 63;
        Wlt[nn * LDP64 + kk] = (__bf16)Wlin[idx];
    }
    __syncthreads();

    // (c) xq = xqpre @ Wlin     (WMMA)
    mm_bf16(Qbf, LDP64, Wlt, LDP64, XQf, 64, 64, 64, 64, wave, 8, lane);
    __syncthreads();

    // (d) sq[i] = (xq[i]+blin)·aq ; sx[j] = x[j]·ax
    if (tid < 64) {
        float a = 0.f, b2 = 0.f;
        for (int c = 0; c < 64; ++c) {
            a  += (XQf[tid * 64 + c] + blin[c]) * attw[c];
            b2 += Xf[tid * 64 + c] * attw[64 + c];
        }
        sq[tid] = a; sx[tid] = b2;
    }
    __syncthreads();

    // (e) masked leaky softmax rows
    if (tid < 64) {
        if (tid < n) {
            float m = -3.4e38f;
            for (int j = 0; j < n; ++j)
                if (A1f[tid * 64 + j] != 0.f) {
                    float l = sq[tid] + sx[j] + ab;
                    l = (l >= 0.f) ? l : 0.2f * l;
                    m = fmaxf(m, l);
                }
            float s = 0.f;
            for (int j = 0; j < 64; ++j) {
                float val = 0.f;
                if (j < n && A1f[tid * 64 + j] != 0.f) {
                    float l = sq[tid] + sx[j] + ab;
                    l = (l >= 0.f) ? l : 0.2f * l;
                    val = __expf(l - m);
                    s += val;
                }
                Sf[tid * 64 + j] = val;
            }
            float inv = 1.f / s;
            for (int j = 0; j < 64; ++j) Sf[tid * 64 + j] *= inv;
        } else {
            for (int j = 0; j < 64; ++j) Sf[tid * 64 + j] = 0.f;
        }
    }
    __syncthreads();
    for (int idx = tid; idx < 4096; idx += 256) {
        int i = idx >> 6, j = idx & 63;
        Sbf[i * LDP64 + j] = (__bf16)Sf[idx];
    }
    __syncthreads();

    // (f) xn = S @ x            (WMMA; x given transposed)
    mm_bf16(Sbf, LDP64, XbfT, LDP64, XNf, 64, 64, 64, 64, wave, 8, lane);
    __syncthreads();

    // (g) matvecs for fitness
    if (tid < 64) {
        float a = leb1[0], b2 = 0.f, c2 = 0.f;
        for (int c = 0; c < 64; ++c) {
            float xv = XNf[tid * 64 + c];
            a  += xv * lew1[c];
            b2 += xv * lew2[c];
            c2 += xv * lew3[c];
        }
        v1[tid] = a; v2[tid] = b2; v3[tid] = c2;
    }
    __syncthreads();
    if (tid < n) {
        float s1 = 0.f, bs = 0.f;
        for (int j = 0; j < n; ++j)
            if (A1f[tid * 64 + j] != 0.f) { s1 += v1[j]; bs += 1.f; }
        float fit = s1 - bs * v2[tid] + v3[tid] + leb3[0];
        fitv[tid] = 1.f / (1.f + __expf(-fit));
    }
    __syncthreads();

    // (i) stable top-k by ranking (ties -> lower index first, like lax.top_k)
    if (tid < n) {
        float fi = fitv[tid];
        int r = 0;
        for (int j = 0; j < n; ++j) {
            float fj = fitv[j];
            if (fj > fi || (fj == fi && j < tid)) ++r;
        }
        if (r < k) { perm[r] = tid; topv[r] = fi; }
    }
    __syncthreads();

    // (j) xo = xn[perm] * topv  -> hout; readout accumulate into xs
    for (int idx = tid; idx < 4096; idx += 256) {
        int r = idx >> 6, c = idx & 63;
        float v = (r < k) ? XNf[perm[r] * 64 + c] * topv[r] : 0.f;
        XQf[idx] = v;                             // xo in LDS
        hout[(size_t)g * 4096 + idx] = v;
    }
    __syncthreads();
    if (tid < 64) {
        float s = 0.f, m = -3.4e38f;
        for (int r = 0; r < k; ++r) {
            float v = XQf[r * 64 + tid];
            s += v; m = fmaxf(m, v);
        }
        xs[(size_t)g * 128 + tid]      += s / (float)k;
        xs[(size_t)g * 128 + 64 + tid] += m;
    }
    __syncthreads();

    // (k) P = S[perm] (k x n, row-major, padded)
    for (int idx = tid; idx < 4096; idx += 256) {
        int r = idx >> 6, j = idx & 63;
        Pbf[r * LDP64 + j] = (__bf16)((r < k) ? Sf[perm[r] * 64 + j] : 0.f);
    }
    __syncthreads();

    // (l) T = A1 @ P^T (B^T of P^T is P itself) ; An = P @ T
    mm_bf16(A1bf, LDP64, Pbf, LDP64, Tf, 64, 64, 64, 64, wave, 8, lane);
    __syncthreads();
    for (int idx = tid; idx < 4096; idx += 256) {
        int s = idx >> 6, j = idx & 63;
        TbfT[s * LDP64 + j] = (__bf16)Tf[j * 64 + s];   // T^T for B operand
    }
    __syncthreads();
    mm_bf16(Pbf, LDP64, TbfT, LDP64, ANf, 64, 64, 64, 64, wave, 8, lane);
    __syncthreads();
    for (int idx = tid; idx < 4096; idx += 256) {
        int r = idx >> 6, s = idx & 63;
        Aout[(size_t)g * 4096 + idx] = (r < k && s < k && r != s) ? ANf[idx] : 0.f;
    }
}

// pred = relu(xs @ l1W + l1b) @ l2W + l2b ; one block per 64 graphs
__global__ __launch_bounds__(256) void head_kernel(
    const float* __restrict__ xs, const float* __restrict__ l1W,
    const float* __restrict__ l1b, const float* __restrict__ l2W,
    const float* __restrict__ l2b, float* __restrict__ pred)
{
    __shared__ __align__(16) __bf16 XSbf[64 * LDP128];
    __shared__ __align__(16) __bf16 W1t[64 * LDP128];   // B^T of l1W
    __shared__ __align__(16) __bf16 T1bf[64 * LDP64];
    __shared__ __align__(16) __bf16 W2t[64 * LDP64];    // B^T of l2W
    __shared__ float  T1f[4096];
    int b = blockIdx.x, tid = threadIdx.x;
    int lane = tid & 31, wave = tid >> 5;

    for (int idx = tid; idx < 64 * 128; idx += 256) {
        int i = idx >> 7, c = idx & 127;
        XSbf[i * LDP128 + c] = (__bf16)xs[(size_t)b * 64 * 128 + idx];
    }
    for (int idx = tid; idx < 128 * 64; idx += 256) {
        int kk = idx >> 6, nn = idx & 63;
        W1t[nn * LDP128 + kk] = (__bf16)l1W[idx];
    }
    for (int idx = tid; idx < 4096; idx += 256) {
        int kk = idx >> 6, nn = idx & 63;
        W2t[nn * LDP64 + kk] = (__bf16)l2W[idx];
    }
    __syncthreads();
    mm_bf16(XSbf, LDP128, W1t, LDP128, T1f, 64, 64, 64, 128, wave, 8, lane);
    __syncthreads();
    for (int idx = tid; idx < 4096; idx += 256) {
        int i = idx >> 6, c = idx & 63;
        T1bf[i * LDP64 + c] = (__bf16)fmaxf(T1f[idx] + l1b[c], 0.f);
    }
    __syncthreads();
    mm_bf16(T1bf, LDP64, W2t, LDP64, T1f, 64, 64, 64, 64, wave, 8, lane);
    __syncthreads();
    for (int idx = tid; idx < 4096; idx += 256)
        pred[(size_t)b * 4096 + idx] = T1f[idx] + l2b[idx & 63];
}

__global__ void finalize_kernel(const float* __restrict__ h, const float* __restrict__ Afin,
                                float* __restrict__ out_h, float* __restrict__ out_A,
                                float* __restrict__ out_emb)
{
    int g = blockIdx.x, tid = threadIdx.x;
    for (int idx = tid; idx < K3_ * HID_; idx += 256)
        out_h[(size_t)g * K3_ * HID_ + idx] = h[(size_t)g * 4096 + idx];
    for (int idx = tid; idx < K3_ * K3_; idx += 256)
        out_A[(size_t)g * K3_ * K3_ + idx] =
            Afin[(size_t)g * 4096 + (size_t)(idx / K3_) * 64 + (idx % K3_)];
    if (tid < HID_) {
        float s = 0.f;
        for (int r = 0; r < K3_; ++r) s += h[(size_t)g * 4096 + r * 64 + tid];
        out_emb[(size_t)g * HID_ + tid] = s / (float)K3_;
    }
}

// ---------------- launcher ----------------
extern "C" void kernel_launch(void* const* d_in, const int* in_sizes, int n_in,
                              void* d_out, int out_size, void* d_ws, size_t ws_size,
                              hipStream_t stream) {
    (void)in_sizes; (void)n_in; (void)out_size; (void)ws_size;
    const float* x     = (const float*)d_in[0];
    const int*   ei    = (const int*)d_in[1];
    const float* W1    = (const float*)d_in[3];
    const float* b1    = (const float*)d_in[4];
    const float* cW    = (const float*)d_in[5];
    const float* cb    = (const float*)d_in[6];
    const float* pWlin = (const float*)d_in[7];
    const float* pblin = (const float*)d_in[8];
    const float* pattw = (const float*)d_in[9];
    const float* pattb = (const float*)d_in[10];
    const float* plew1 = (const float*)d_in[11];
    const float* pleb1 = (const float*)d_in[12];
    const float* plew2 = (const float*)d_in[13];
    const float* plew3 = (const float*)d_in[14];
    const float* pleb3 = (const float*)d_in[15];
    const float* l1W   = (const float*)d_in[16];
    const float* l1b   = (const float*)d_in[17];
    const float* l2W   = (const float*)d_in[18];
    const float* l2b   = (const float*)d_in[19];

    float* ws  = (float*)d_ws;
    float* A0  = ws;                         // G*4096
    float* A1b = ws + (size_t)G_ * 4096;     // G*4096
    float* h0  = ws + (size_t)2 * G_ * 4096;
    float* h1  = ws + (size_t)3 * G_ * 4096;
    float* xsb = ws + (size_t)4 * G_ * 4096; // G*128
    float* out = (float*)d_out;

    const int E = G_ * EPG_;
    const int OFF_H = G_ * OUTC_;
    const int OFF_A = OFF_H + G_ * K3_ * HID_;
    const int OFF_E = OFF_A + G_ * K3_ * K3_;

    zero_f32<<<(G_ * 4096 + 255) / 256, 256, 0, stream>>>(A0, G_ * 4096);
    zero_f32<<<(G_ * 128 + 255) / 256, 256, 0, stream>>>(xsb, G_ * 128);
    scatter_adj<<<(E + 255) / 256, 256, 0, stream>>>(ei, A0, E);

    // stage 1
    gcn_kernel<<<G_, 256, 0, stream>>>(x, INC_, A0, 0, W1, b1, h0, N0_);
    pool_kernel<<<G_, 256, 0, stream>>>(h0, A0, pWlin, pblin, pattw, pattb,
                                        plew1, pleb1, plew2, plew3, pleb3,
                                        h1, A1b, xsb, N0_, K1_);
    // stage 2
    gcn_kernel<<<G_, 256, 0, stream>>>(h1, HID_, A1b, 1, cW, cb, h0, K1_);
    pool_kernel<<<G_, 256, 0, stream>>>(h0, A1b, pWlin + 4096, pblin + 64,
                                        pattw + 128, pattb + 1, plew1 + 64, pleb1 + 1,
                                        plew2 + 64, plew3 + 64, pleb3 + 1,
                                        h1, A0, xsb, K1_, K2_);
    // stage 3
    gcn_kernel<<<G_, 256, 0, stream>>>(h1, HID_, A0, 1, cW + 4096, cb + 64, h0, K2_);
    pool_kernel<<<G_, 256, 0, stream>>>(h0, A0, pWlin + 2 * 4096, pblin + 128,
                                        pattw + 256, pattb + 2, plew1 + 128, pleb1 + 2,
                                        plew2 + 128, plew3 + 128, pleb3 + 2,
                                        h1, A1b, xsb, K2_, K3_);

    head_kernel<<<G_ / 64, 256, 0, stream>>>(xsb, l1W, l1b, l2W, l2b, out);
    finalize_kernel<<<G_, 256, 0, stream>>>(h1, A1b, out + OFF_H, out + OFF_A, out + OFF_E);
}
